// DDGCRN_56075093017381
// MI455X (gfx1250) — matrix operand order repaired and checked by reading, not compiled, not run on hardware
//
#include <hip/hip_runtime.h>
#include <hip/hip_bf16.h>

// ---------------------------------------------------------------------------
// DDGCRN on MI455X (gfx1250): WMMA-f16 GEMMs for the per-timestep matvecs,
// f32 recurrent state with an f16 mirror for the A-operand, deterministic
// tree reductions for the cross-node "pooled" sum, weights pre-packed into
// the CDNA5 WMMA B-operand lane layout.  One wave = one (b, nodeTile) for one
// matrix (selected by blockIdx parity -> scalar-uniform branch), looping over
// all col tiles so the A operand and x row-broadcast are reused.
// ---------------------------------------------------------------------------

#define BB   16
#define TT   12
#define NN   8600
#define HIDD 64
#define G2   128           // 2*HID
#define NT   538           // ceil(8600/16)
#define NPAD (NT * 16)     // 8608

typedef __attribute__((ext_vector_type(16))) _Float16 v16h;
typedef __attribute__((ext_vector_type(8)))  _Float16 v8h;
typedef __attribute__((ext_vector_type(8)))  float    v8f;

struct P {
    // inputs (f32)
    const float *src;
    const float *gaW, *gab, *gw, *gb, *gnw, *gaw, *gaffw, *gaffb;
    const float *uaW, *uab, *uw, *ub, *unw, *uaw, *uaffw, *uaffb;
    const float *cw, *cb;
    // workspace
    float *state;            // [B][NPAD][64] f32 recurrent state
    float *res_g;            // [B][NPAD][128] gate align pre-activation
    float *res_u;            // [B][NPAD][64]  upd  align pre-activation
    float *part_g;           // [B][NT][128] per-node-tile pool partials (gate)
    float *part_u;           // [B][NT][64]
    float *pool_g;           // [B][128]
    float *pool_u;           // [B][64]
    float *src2;             // [B][T][N] layer-2 input (source - source1)
    _Float16 *packW;         // packed WMMA B-operand weights (48 tiles x 1024 half)
    _Float16 *state_h;       // f16 mirror of state    (A operand, gate GEMM)
    _Float16 *zst_h;         // f16 z*state            (A operand, upd GEMM)
    float *out;              // d_out [B][T][N]
};

__device__ __forceinline__ float sigmoidf_(float x) {
    return 1.0f / (1.0f + __expf(-x));
}

// Load the K=64 A operand (two 16x16x32 kstep fragments) for row = lane&15.
// A layout (16-bit A 16x32): VGPRs 0-3 hold K = 8h+0..7, VGPRs 4-7 hold
// K = 16+8h+0..7 -> two contiguous 8-half runs per kstep.
__device__ __forceinline__ void load_a64(const _Float16 *ap, int h,
                                         v16h &a0, v16h &a1) {
    v8h x0 = *(const v8h *)(ap + 8 * h);
    v8h x1 = *(const v8h *)(ap + 16 + 8 * h);
    v8h x2 = *(const v8h *)(ap + 32 + 8 * h);
    v8h x3 = *(const v8h *)(ap + 48 + 8 * h);
#pragma unroll
    for (int i = 0; i < 8; ++i) {
        a0[i] = x0[i]; a0[8 + i] = x1[i];
        a1[i] = x2[i]; a1[8 + i] = x3[i];
    }
}

// One 16x16 tile: C initialized to bias, two K=32 WMMAs.
__device__ __forceinline__ v8f tile_gemm(const v16h &a0, const v16h &a1,
                                         const _Float16 *bt, int lane, float bc) {
    v16h b0 = *(const v16h *)(bt + (size_t)lane * 16);
    v16h b1 = *(const v16h *)(bt + 512 + (size_t)lane * 16);
    v8f c;
#pragma unroll
    for (int i = 0; i < 8; ++i) c[i] = bc;
    c = __builtin_amdgcn_wmma_f32_16x16x32_f16(false, a0, false, b0,
                                               (short)0, c, false, false);
    c = __builtin_amdgcn_wmma_f32_16x16x32_f16(false, a1, false, b1,
                                               (short)0, c, false, false);
    return c;
}

// Branch-free per-lane x fetch + broadcast of the 8 output-row x values.
__device__ __forceinline__ void fetch_xrow(const float *xs, size_t xbase,
                                           int nt, int lane, int h,
                                           float (&xrow)[8]) {
    int node = nt * 16 + (lane & 15);
    int safe = node < NN ? node : (NN - 1);
    float xl = xs[xbase + safe];
    xl = node < NN ? xl : 0.0f;          // v_cndmask, no exec branching
#pragma unroll
    for (int r = 0; r < 8; ++r)
        xrow[r] = __shfl(xl, 8 * h + r, 32);
}

// ---------------------------------------------------------------------------
// Pack f32 weights (rows 1..64 = state part) into WMMA B-operand layout:
// B 32x16 f16: lane supplies col N = lane&15; element e of v16h holds
// K = 8*(lane>>4) + e (e<8), 16 + 8*(lane>>4) + (e-8) (e>=8).  Tile ids per layer: [0..7] gate_align, [8..15]
// gate_w, [16..19] upd_align, [20..23] upd_w.  Each tile: 2 ksteps * 512 h.
// ---------------------------------------------------------------------------
__global__ void k_pack(P p) {
    int gid = blockIdx.x * blockDim.x + threadIdx.x;
    if (gid >= 49152) return;
    int e    = gid & 15;
    int lane = (gid >> 4) & 31;
    int s    = (gid >> 9) & 1;
    int tile = gid >> 10;
    int l = tile / 24, sub = tile % 24;
    int h = lane >> 4, q = lane & 15;
    int krow = 1 + s * 32 + 8 * h + e + (e >= 8 ? 8 : 0);     // +1 skips the x row (row 0)
    float val;
    if (sub < 16) {
        int isw = sub >> 3, ctl = sub & 7;
        const float *W = isw ? p.gw : p.gaW;
        val = W[((size_t)l * 65 + krow) * G2 + ctl * 16 + q];
    } else {
        int s2 = sub - 16, isw = s2 >> 2, ctl = s2 & 3;
        const float *W = isw ? p.uw : p.uaW;
        val = W[((size_t)l * 65 + krow) * HIDD + ctl * 16 + q];
    }
    p.packW[gid] = (_Float16)val;
}

__global__ void k_zero_state(P p) {
    size_t tid = (size_t)blockIdx.x * blockDim.x + threadIdx.x;
    if (tid >= (size_t)BB * NPAD * HIDD) return;
    p.state[tid]   = 0.0f;
    p.state_h[tid] = (_Float16)0.0f;
}

// ---------------------------------------------------------------------------
// Gate GEMM.  blockIdx parity selects the matrix (SGPR -> uniform branch):
// even blocks = alignW (store res_g unconditionally; padded rows are masked
// downstream), odd blocks = w (relu + pooled partials, stores hoisted out of
// the col-tile loop).  8608 waves per matrix, 4 waves/block.
// ---------------------------------------------------------------------------
__global__ void k_gate_gemm(P p, int l, int t) {
    const int isw  = blockIdx.x & 1;                     // scalar-uniform
    const int unit = (blockIdx.x >> 1) * 4 + (threadIdx.x >> 5); // 0..8607
    const int lane = threadIdx.x & 31;
    const int nt = unit % NT;
    const int b  = unit / NT;
    const int h = lane >> 4, q = lane & 15;

    v16h a0, a1;
    load_a64(p.state_h + (((size_t)b * NPAD) + nt * 16 + q) * HIDD, h, a0, a1);

    const float *xs = l ? p.src2 : p.src;
    size_t xbase = ((size_t)b * TT + t) * NN;
    float xrow[8];
    fetch_xrow(xs, xbase, nt, lane, h, xrow);

    const int nodeBase = nt * 16 + 8 * h;

    if (isw == 0) {
        const float *Wf = p.gaW + (size_t)l * 65 * G2;
        const float *bb = p.gab + l * G2;
        const _Float16 *btb = p.packW + (size_t)(l * 24) * 1024;
        float *resb = p.res_g + ((size_t)b * NPAD + nodeBase) * G2;
#pragma unroll
        for (int ctl = 0; ctl < 8; ++ctl) {
            int col = ctl * 16 + q;
            v8f c = tile_gemm(a0, a1, btb + (size_t)ctl * 1024, lane, bb[col]);
            float w0 = Wf[col];
#pragma unroll
            for (int r = 0; r < 8; ++r)
                resb[(size_t)r * G2 + col] = c[r] + xrow[r] * w0;
        }
    } else {
        float vm[8];
#pragma unroll
        for (int r = 0; r < 8; ++r) vm[r] = (nodeBase + r < NN) ? 1.0f : 0.0f;
        const float *Wf = p.gw + (size_t)l * 65 * G2;
        const float *bb = p.gb + l * G2;
        const _Float16 *btb = p.packW + (size_t)(l * 24 + 8) * 1024;
        float ps[8];
#pragma unroll
        for (int ctl = 0; ctl < 8; ++ctl) {
            int col = ctl * 16 + q;
            v8f c = tile_gemm(a0, a1, btb + (size_t)ctl * 1024, lane, bb[col]);
            float w0 = Wf[col];
            float s = 0.0f;
#pragma unroll
            for (int r = 0; r < 8; ++r) {
                float d = c[r] + xrow[r] * w0;
                float hv = d > 0.0f ? d : 0.0f;      // relu
                s += hv * vm[r];                     // mask padded rows
            }
            ps[ctl] = s + __shfl_xor(s, 16, 32);     // combine row halves
        }
        if (lane < 16) {                             // single exec toggle
            float *pg = p.part_g + (((size_t)b * NT) + nt) * G2 + q;
#pragma unroll
            for (int ctl = 0; ctl < 8; ++ctl) pg[ctl * 16] = ps[ctl];
        }
    }
}

// Deterministic fixed-order reduction of per-tile partials -> pooled[b][c].
__global__ void k_reduce(const float *part, float *pool, int cols) {
    int tid = blockIdx.x * blockDim.x + threadIdx.x;
    if (tid >= BB * cols) return;
    int c = tid % cols, b = tid / cols;
    const float *pp = part + (size_t)b * NT * cols + c;
    float s = 0.0f;
    for (int nt = 0; nt < NT; ++nt) s += pp[(size_t)nt * cols];
    pool[tid] = s;
}

// z = sigmoid(res_g + affw*(pooled*scale) + affb); zst = f16(z * state).
// Written for all NPAD nodes (padded -> 0) so the upd GEMM A-loads need no mask.
__global__ void k_zstate(P p, int l) {
    size_t tid = (size_t)blockIdx.x * blockDim.x + threadIdx.x;
    if (tid >= (size_t)BB * NPAD * HIDD) return;
    int k = (int)(tid & 63);
    int n = (int)((tid >> 6) % NPAD);
    int b = (int)(tid / ((size_t)NPAD * HIDD));
    _Float16 out = (_Float16)0.0f;
    if (n < NN) {
        size_t nl = (size_t)l * NN + n;
        float scale = p.gaw[nl] * p.gnw[nl];
        float g = p.res_g[((size_t)b * NPAD + n) * G2 + k]
                + p.gaffw[nl * G2 + k] * (p.pool_g[b * G2 + k] * scale)
                + p.gaffb[nl * G2 + k];
        out = (_Float16)(sigmoidf_(g) * p.state[((size_t)b * NPAD + n) * HIDD + k]);
    }
    p.zst_h[tid] = out;
}

// Update GEMM: same structure as the gate GEMM, 4 col tiles per matrix.
__global__ void k_upd_gemm(P p, int l, int t) {
    const int isw  = blockIdx.x & 1;                     // scalar-uniform
    const int unit = (blockIdx.x >> 1) * 4 + (threadIdx.x >> 5);
    const int lane = threadIdx.x & 31;
    const int nt = unit % NT;
    const int b  = unit / NT;
    const int h = lane >> 4, q = lane & 15;

    v16h a0, a1;
    load_a64(p.zst_h + (((size_t)b * NPAD) + nt * 16 + q) * HIDD, h, a0, a1);

    const float *xs = l ? p.src2 : p.src;
    size_t xbase = ((size_t)b * TT + t) * NN;
    float xrow[8];
    fetch_xrow(xs, xbase, nt, lane, h, xrow);

    const int nodeBase = nt * 16 + 8 * h;

    if (isw == 0) {
        const float *Wf = p.uaW + (size_t)l * 65 * HIDD;
        const float *bb = p.uab + l * HIDD;
        const _Float16 *btb = p.packW + (size_t)(l * 24 + 16) * 1024;
        float *resb = p.res_u + ((size_t)b * NPAD + nodeBase) * HIDD;
#pragma unroll
        for (int ctl = 0; ctl < 4; ++ctl) {
            int col = ctl * 16 + q;
            v8f c = tile_gemm(a0, a1, btb + (size_t)ctl * 1024, lane, bb[col]);
            float w0 = Wf[col];
#pragma unroll
            for (int r = 0; r < 8; ++r)
                resb[(size_t)r * HIDD + col] = c[r] + xrow[r] * w0;
        }
    } else {
        float vm[8];
#pragma unroll
        for (int r = 0; r < 8; ++r) vm[r] = (nodeBase + r < NN) ? 1.0f : 0.0f;
        const float *Wf = p.uw + (size_t)l * 65 * HIDD;
        const float *bb = p.ub + l * HIDD;
        const _Float16 *btb = p.packW + (size_t)(l * 24 + 20) * 1024;
        float ps[4];
#pragma unroll
        for (int ctl = 0; ctl < 4; ++ctl) {
            int col = ctl * 16 + q;
            v8f c = tile_gemm(a0, a1, btb + (size_t)ctl * 1024, lane, bb[col]);
            float w0 = Wf[col];
            float s = 0.0f;
#pragma unroll
            for (int r = 0; r < 8; ++r) {
                float d = c[r] + xrow[r] * w0;
                float hv = d > 0.0f ? d : 0.0f;
                s += hv * vm[r];
            }
            ps[ctl] = s + __shfl_xor(s, 16, 32);
        }
        if (lane < 16) {
            float *pu = p.part_u + (((size_t)b * NT) + nt) * HIDD + q;
#pragma unroll
            for (int ctl = 0; ctl < 4; ++ctl) pu[ctl * 16] = ps[ctl];
        }
    }
}

// r = sigmoid(gate cols 64..127); hc = tanh(upd); state = r*state + (1-r)*hc.
__global__ void k_update(P p, int l) {
    size_t tid = (size_t)blockIdx.x * blockDim.x + threadIdx.x;
    if (tid >= (size_t)BB * NN * HIDD) return;
    int k = (int)(tid & 63);
    size_t bn = tid >> 6;
    int n = (int)(bn % NN), b = (int)(bn / NN);
    size_t nl = (size_t)l * NN + n;
    size_t ib = (size_t)b * NPAD + n;

    float gscale = p.gaw[nl] * p.gnw[nl];
    float uscale = p.uaw[nl] * p.unw[nl];
    float rg = p.res_g[ib * G2 + 64 + k]
             + p.gaffw[nl * G2 + 64 + k] * (p.pool_g[b * G2 + 64 + k] * gscale)
             + p.gaffb[nl * G2 + 64 + k];
    float r  = sigmoidf_(rg);
    float hu = p.res_u[ib * HIDD + k]
             + p.uaffw[nl * HIDD + k] * (p.pool_u[b * HIDD + k] * uscale)
             + p.uaffb[nl * HIDD + k];
    float hc = tanhf(hu);
    float sn = r * p.state[ib * HIDD + k] + (1.0f - r) * hc;
    p.state[ib * HIDD + k]   = sn;
    p.state_h[ib * HIDD + k] = (_Float16)sn;    // f16 mirror for next gate GEMM
}

// Heads: phase 0 -> out = h1 . cw[0] + cb[0]; src2 = src - (h1 . cw[1] + cb[1]).
//        phase 1 -> out += h2 . cw[2] + cb[2].
__global__ void k_head(P p, int phase) {
    int tid = blockIdx.x * blockDim.x + threadIdx.x;
    if (tid >= BB * NN) return;
    int n = tid % NN, b = tid / NN;
    float hreg[HIDD];
    const float *hp = p.state + ((size_t)b * NPAD + n) * HIDD;
#pragma unroll
    for (int j = 0; j < HIDD; ++j) hreg[j] = hp[j];
    if (phase == 0) {
        for (int t = 0; t < TT; ++t) {
            const float *w0 = p.cw + (0 * TT + t) * HIDD;
            const float *w1 = p.cw + (1 * TT + t) * HIDD;
            float d0 = 0.0f, d1 = 0.0f;
#pragma unroll
            for (int j = 0; j < HIDD; ++j) { d0 += hreg[j] * w0[j]; d1 += hreg[j] * w1[j]; }
            size_t oi = ((size_t)b * TT + t) * NN + n;
            p.out[oi]  = d0 + p.cb[t];
            p.src2[oi] = p.src[oi] - (d1 + p.cb[TT + t]);
        }
    } else {
        for (int t = 0; t < TT; ++t) {
            const float *w2 = p.cw + (2 * TT + t) * HIDD;
            float d2 = 0.0f;
#pragma unroll
            for (int j = 0; j < HIDD; ++j) d2 += hreg[j] * w2[j];
            size_t oi = ((size_t)b * TT + t) * NN + n;
            p.out[oi] += d2 + p.cb[2 * TT + t];
        }
    }
}

extern "C" void kernel_launch(void *const *d_in, const int *in_sizes, int n_in,
                              void *d_out, int out_size, void *d_ws, size_t ws_size,
                              hipStream_t stream) {
    P p;
    p.src   = (const float *)d_in[0];
    p.gaW   = (const float *)d_in[1];
    p.gab   = (const float *)d_in[2];
    p.gw    = (const float *)d_in[3];
    p.gb    = (const float *)d_in[4];
    p.gnw   = (const float *)d_in[5];
    p.gaw   = (const float *)d_in[6];
    p.gaffw = (const float *)d_in[7];
    p.gaffb = (const float *)d_in[8];
    p.uaW   = (const float *)d_in[9];
    p.uab   = (const float *)d_in[10];
    p.uw    = (const float *)d_in[11];
    p.ub    = (const float *)d_in[12];
    p.unw   = (const float *)d_in[13];
    p.uaw   = (const float *)d_in[14];
    p.uaffw = (const float *)d_in[15];
    p.uaffb = (const float *)d_in[16];
    p.cw    = (const float *)d_in[17];
    p.cb    = (const float *)d_in[18];
    p.out   = (float *)d_out;

    float *ws = (float *)d_ws;
    size_t o = 0;
    p.state  = ws + o; o += (size_t)BB * NPAD * HIDD;
    p.res_g  = ws + o; o += (size_t)BB * NPAD * G2;
    p.res_u  = ws + o; o += (size_t)BB * NPAD * HIDD;
    p.part_g = ws + o; o += (size_t)BB * NT * G2;
    p.part_u = ws + o; o += (size_t)BB * NT * HIDD;
    p.pool_g = ws + o; o += (size_t)BB * G2;
    p.pool_u = ws + o; o += (size_t)BB * HIDD;
    p.src2   = ws + o; o += (size_t)BB * TT * NN;
    p.packW   = (_Float16 *)(ws + o); o += 49152 / 2;
    p.state_h = (_Float16 *)(ws + o); o += (size_t)BB * NPAD * HIDD / 2;
    p.zst_h   = (_Float16 *)(ws + o); o += (size_t)BB * NPAD * HIDD / 2;

    const int ZB = (int)(((size_t)BB * NPAD * HIDD) / 256);   // 34432, exact
    const int UB = (int)(((size_t)BB * NN * HIDD) / 256);     // 34400, exact
    const int GB = 2 * (BB * NT / 4);                         // 4304 blocks, exact

    k_pack<<<192, 256, 0, stream>>>(p);

    for (int l = 0; l < 2; ++l) {
        k_zero_state<<<ZB, 256, 0, stream>>>(p);
        for (int t = 0; t < TT; ++t) {
            k_gate_gemm<<<GB, 128, 0, stream>>>(p, l, t);
            k_reduce<<<(BB * G2) / 256, 256, 0, stream>>>(p.part_g, p.pool_g, G2);
            k_zstate<<<ZB, 256, 0, stream>>>(p, l);
            k_upd_gemm<<<GB, 128, 0, stream>>>(p, l, t);
            k_reduce<<<(BB * HIDD) / 256, 256, 0, stream>>>(p.part_u, p.pool_u, HIDD);
            k_update<<<UB, 256, 0, stream>>>(p, l);
        }
        k_head<<<(BB * NN + 255) / 256, 256, 0, stream>>>(p, l);
    }
}
